// ContrastiveLoss_1331439862529
// MI455X (gfx1250) — matrix operand order repaired and checked by reading, not compiled, not run on hardware
//
#include <hip/hip_runtime.h>
#include <hip/hip_bf16.h>

typedef __attribute__((ext_vector_type(16))) _Float16     v16h;
typedef __attribute__((ext_vector_type(8)))  float        v8f;
typedef __attribute__((ext_vector_type(4)))  unsigned int u32x4;

#define NB     1024      // batch size B
#define NROWS  2048      // 2B
#define DDIM   16384     // 128*128 flat
#define INV_T  2.0f      // 1 / temperature(0.5)

// ---------------------------------------------------------------------------
// Phase 1: per-column (dim=1) L2 normalize + flat-row renormalize -> fp16 rn.
// One block of 256 threads per row. Element (m,n) lives at m*128+n, so a
// stride-256 walk keeps each thread pinned to a single column n = t & 127.
// ---------------------------------------------------------------------------
__global__ __launch_bounds__(256) void nt_xent_normalize(
    const float* __restrict__ emb_i, const float* __restrict__ emb_j,
    _Float16* __restrict__ rn) {
  const int r = blockIdx.x;
  const float* __restrict__ src =
      (r < NB) ? (emb_i + (size_t)r * DDIM) : (emb_j + (size_t)(r - NB) * DDIM);
  __shared__ float s_part[256];
  __shared__ float s_scale[128];
  __shared__ float s_rowscale;
  const int t = threadIdx.x;

  float acc = 0.0f;
  for (int idx = t; idx < DDIM; idx += 256) {
    float v = src[idx];
    acc += v * v;                       // column n = t & 127 only
  }
  s_part[t] = acc;
  __syncthreads();
  if (t < 128) {
    float s  = s_part[t] + s_part[t + 128];          // column sum-of-squares
    float cs = 1.0f / fmaxf(sqrtf(s), 1e-12f);       // F.normalize eps
    s_scale[t] = cs;
    s_part[t]  = s * cs * cs;                        // contribution to flat |.|^2
  }
  __syncthreads();
  if (t == 0) {
    float fs = 0.0f;
    for (int n = 0; n < 128; ++n) fs += s_part[n];
    s_rowscale = 1.0f / fmaxf(sqrtf(fs), 1e-8f);     // cosine-sim eps
  }
  __syncthreads();
  const float rowScale = s_rowscale;
  _Float16* __restrict__ dst = rn + (size_t)r * DDIM;
  for (int idx = t; idx < DDIM; idx += 256)
    dst[idx] = (_Float16)(src[idx] * s_scale[idx & 127] * rowScale);
}

// ---------------------------------------------------------------------------
// Phase 2: sim = rn @ rn^T fused with exp/mask epilogue.
// Block = 256 threads = 8 wave32, computes a 128x128 tile of sim.
// Waves laid out 4 (rows of 32) x 2 (cols of 64); each wave owns 2x4 WMMA
// accumulator tiles of 16x16 and runs K=16384 in steps of 32 with
// v_wmma_f32_16x16x32_f16. Fragments loaded straight from L2-resident rn
// using the CDNA5 ISA VGPR layouts:
//   A 16x32 f16 : lane L -> M=L%16; halves 0..7 = K base..base+7,
//                 halves 8..15 = K base+16..base+23, base = (L/16)*8
//   B 32x16 f16 : lane L -> N=L%16; same K layout as A
//   C 16x16 f32 : lane L -> N=L%16; vgpr v -> M = v + 8*(L/16)
// ---------------------------------------------------------------------------
__global__ __launch_bounds__(256) void nt_xent_gemm(
    const _Float16* __restrict__ rn,
    float* __restrict__ partial,      // [NROWS][16] row-sum partials per jblock
    float* __restrict__ pos) {        // [NROWS] positive similarities
  const int jb = blockIdx.x;          // 0..15 column block
  const int ib = blockIdx.y;          // 0..15 row block
  const int i0 = ib << 7;
  const int j0 = jb << 7;
  const int wave   = threadIdx.x >> 5;
  const int lane   = threadIdx.x & 31;
  const int wave_m = wave >> 1;       // 0..3 -> 32 rows each
  const int wave_n = wave & 1;        // 0..1 -> 64 cols each
  const int lmod   = lane & 15;
  const int lhi    = lane >> 4;

  v8f acc[2][4] = {};

  const _Float16* __restrict__ pa0 =
      rn + (size_t)(i0 + wave_m * 32 + 0  + lmod) * DDIM + lhi * 8;
  const _Float16* __restrict__ pa1 =
      rn + (size_t)(i0 + wave_m * 32 + 16 + lmod) * DDIM + lhi * 8;
  const _Float16* __restrict__ pb0 =
      rn + (size_t)(j0 + wave_n * 64 + lmod) * DDIM + lhi * 8;
  const _Float16* __restrict__ pb1 = pb0 + (size_t)16 * DDIM;
  const _Float16* __restrict__ pb2 = pb0 + (size_t)32 * DDIM;
  const _Float16* __restrict__ pb3 = pb0 + (size_t)48 * DDIM;

  union Frag { v16h h; u32x4 u[2]; };

  for (int k = 0; k < DDIM; k += 32) {
    Frag a0, a1, b0, b1, b2, b3;
    a0.u[0] = *(const u32x4*)(pa0 + k);
    a0.u[1] = *(const u32x4*)(pa0 + k + 16);
    a1.u[0] = *(const u32x4*)(pa1 + k);
    a1.u[1] = *(const u32x4*)(pa1 + k + 16);
    b0.u[0] = *(const u32x4*)(pb0 + k);
    b0.u[1] = *(const u32x4*)(pb0 + k + 16);
    b1.u[0] = *(const u32x4*)(pb1 + k);
    b1.u[1] = *(const u32x4*)(pb1 + k + 16);
    b2.u[0] = *(const u32x4*)(pb2 + k);
    b2.u[1] = *(const u32x4*)(pb2 + k + 16);
    b3.u[0] = *(const u32x4*)(pb3 + k);
    b3.u[1] = *(const u32x4*)(pb3 + k + 16);

    acc[0][0] = __builtin_amdgcn_wmma_f32_16x16x32_f16(false, a0.h, false, b0.h, (short)0, acc[0][0], false, false);
    acc[0][1] = __builtin_amdgcn_wmma_f32_16x16x32_f16(false, a0.h, false, b1.h, (short)0, acc[0][1], false, false);
    acc[0][2] = __builtin_amdgcn_wmma_f32_16x16x32_f16(false, a0.h, false, b2.h, (short)0, acc[0][2], false, false);
    acc[0][3] = __builtin_amdgcn_wmma_f32_16x16x32_f16(false, a0.h, false, b3.h, (short)0, acc[0][3], false, false);
    acc[1][0] = __builtin_amdgcn_wmma_f32_16x16x32_f16(false, a1.h, false, b0.h, (short)0, acc[1][0], false, false);
    acc[1][1] = __builtin_amdgcn_wmma_f32_16x16x32_f16(false, a1.h, false, b1.h, (short)0, acc[1][1], false, false);
    acc[1][2] = __builtin_amdgcn_wmma_f32_16x16x32_f16(false, a1.h, false, b2.h, (short)0, acc[1][2], false, false);
    acc[1][3] = __builtin_amdgcn_wmma_f32_16x16x32_f16(false, a1.h, false, b3.h, (short)0, acc[1][3], false, false);
  }

  // ---- epilogue: positives + masked exp row sums ----
  __shared__ float rowsum[128];
  if (threadIdx.x < 128) rowsum[threadIdx.x] = 0.0f;
  __syncthreads();

#pragma unroll
  for (int mt = 0; mt < 2; ++mt) {
#pragma unroll
    for (int v = 0; v < 8; ++v) {
      const int i = i0 + wave_m * 32 + mt * 16 + lhi * 8 + v;
      const int jp = (i < NB) ? (i + NB) : (i - NB);   // positive partner
      float esum = 0.0f;
#pragma unroll
      for (int nt = 0; nt < 4; ++nt) {
        const int j = j0 + wave_n * 64 + nt * 16 + lmod;
        const float s = acc[mt][nt][v];
        if (j == jp) pos[i] = s;                        // written exactly once
        esum += (i == j) ? 0.0f : __expf(s * INV_T);    // mask diagonal
      }
      // reduce across the 16 lanes of this half-wave (one row each half)
      esum += __shfl_xor(esum, 1, 32);
      esum += __shfl_xor(esum, 2, 32);
      esum += __shfl_xor(esum, 4, 32);
      esum += __shfl_xor(esum, 8, 32);
      if (lmod == 0)  // exactly 2 commutative adds per row (wave_n=0,1)
        atomicAdd(&rowsum[i - i0], esum);
    }
  }
  __syncthreads();
  if (threadIdx.x < 128)
    partial[(size_t)(i0 + threadIdx.x) * 16 + jb] = rowsum[threadIdx.x];
}

// ---------------------------------------------------------------------------
// Phase 3: loss = (1/2B) * sum_i [ log(sum_b partial[i][b]) - pos[i]/T ]
// ---------------------------------------------------------------------------
__global__ __launch_bounds__(256) void nt_xent_loss(
    const float* __restrict__ partial, const float* __restrict__ pos,
    float* __restrict__ out) {
  __shared__ float red[256];
  float acc = 0.0f;
  for (int i = threadIdx.x; i < NROWS; i += 256) {
    float denom = 0.0f;
#pragma unroll
    for (int b = 0; b < 16; ++b) denom += partial[(size_t)i * 16 + b];
    acc += __logf(denom) - pos[i] * INV_T;
  }
  red[threadIdx.x] = acc;
  __syncthreads();
  for (int s = 128; s > 0; s >>= 1) {
    if (threadIdx.x < s) red[threadIdx.x] += red[threadIdx.x + s];
    __syncthreads();
  }
  if (threadIdx.x == 0) out[0] = red[0] / (float)NROWS;
}

// ---------------------------------------------------------------------------
extern "C" void kernel_launch(void* const* d_in, const int* in_sizes, int n_in,
                              void* d_out, int out_size, void* d_ws, size_t ws_size,
                              hipStream_t stream) {
  (void)in_sizes; (void)n_in; (void)out_size; (void)ws_size;
  const float* emb_i = (const float*)d_in[0];
  const float* emb_j = (const float*)d_in[1];
  float* out = (float*)d_out;

  char* ws = (char*)d_ws;
  _Float16* rn   = (_Float16*)ws;                               // 64 MB
  float* partial = (float*)(ws + (size_t)NROWS * DDIM * 2);     // 128 KB
  float* pos     = partial + (size_t)NROWS * 16;                // 8 KB

  nt_xent_normalize<<<NROWS, 256, 0, stream>>>(emb_i, emb_j, rn);
  dim3 grid(16, 16);
  nt_xent_gemm<<<grid, 256, 0, stream>>>(rn, partial, pos);
  nt_xent_loss<<<1, 256, 0, stream>>>(partial, pos, out);
}
